// KNRM_60413009985868
// MI455X (gfx1250) — matrix operand order repaired and checked by reading, not compiled, not run on hardware
//
#include <hip/hip_runtime.h>

typedef __attribute__((ext_vector_type(16))) _Float16 v16h;
typedef __attribute__((ext_vector_type(8)))  _Float16 v8h;
typedef __attribute__((ext_vector_type(8)))  float    v8f;

#define QN    32      // queries per batch
#define DN    180     // docs per batch
#define DPAD  192     // padded to 12 N-tiles of 16
#define EN    300     // embedding dim
#define EPAD  320     // padded to 10 K-steps of 32
#define NK    11      // gaussian kernels
#define NTHREADS 256  // 8 wave32

// mu_k and -0.5/sigma_k^2 from kernel_mus/kernel_sigmas(11)
__constant__ float MU_C[NK] = {1.0f, 0.9f, 0.7f, 0.5f, 0.3f, 0.1f,
                               -0.1f, -0.3f, -0.5f, -0.7f, -0.9f};
__constant__ float CF_C[NK] = {-5.0e7f, -50.0f, -50.0f, -50.0f, -50.0f, -50.0f,
                               -50.0f, -50.0f, -50.0f, -50.0f, -50.0f};

__global__ __launch_bounds__(NTHREADS)
void knrm_fused(const int* __restrict__ query,
                const int* __restrict__ doc,
                const float* __restrict__ emb,
                const float* __restrict__ w_out,
                float* __restrict__ out)
{
    extern __shared__ __align__(32) char smem[];
    _Float16* dB = (_Float16*)smem;                              // [DPAD][EPAD] (B^T rows)
    _Float16* qA = (_Float16*)(smem + (size_t)DPAD * EPAD * 2);  // [QN][EPAD]
    float* ks    = (float*)(smem + (size_t)(DPAD + QN) * EPAD * 2); // [QN][NK]
    float* qm    = ks + QN * NK;    // [QN]
    float* dmk   = qm + QN;         // [DPAD]
    float* part  = dmk + DPAD;      // [16]

    const int tid  = threadIdx.x;
    const int w    = tid >> 5;   // wave id 0..7
    const int lane = tid & 31;
    const int b    = blockIdx.x;

    for (int i = tid; i < QN * NK; i += NTHREADS) ks[i] = 0.0f;

    // ---------------- Phase 1: gather + L2-normalize -> f16 LDS ----------------
    for (int row = w; row < QN + DPAD; row += 8) {
        _Float16* dst;
        int srcIdx;
        float maskv = 0.0f;
        if (row < QN) {
            srcIdx = query[b * QN + row];
            maskv  = (srcIdx > 0) ? 1.0f : 0.0f;
            dst    = qA + row * EPAD;
        } else {
            const int n = row - QN;
            dst = dB + n * EPAD;
            if (n < DN) {
                srcIdx = doc[b * DN + n];
                maskv  = (srcIdx > 0) ? 1.0f : 0.0f;
            } else {
                srcIdx = -1;  // zero-fill padded doc rows
            }
        }
        float vals[10];
        float ss = 0.0f;
        if (srcIdx >= 0) {
            const float* src = emb + (size_t)srcIdx * EN;
            #pragma unroll
            for (int i = 0; i < 10; ++i) {
                const int e = lane + 32 * i;
                const float v = (e < EN) ? src[e] : 0.0f;
                vals[i] = v;
                ss += v * v;
            }
        } else {
            #pragma unroll
            for (int i = 0; i < 10; ++i) vals[i] = 0.0f;
        }
        // wave32 reduction of sum of squares
        #pragma unroll
        for (int off = 16; off >= 1; off >>= 1) ss += __shfl_xor(ss, off, 32);
        float scale = (srcIdx >= 0) ? (1.0f / (sqrtf(ss) + 1e-13f)) : 0.0f;
        #pragma unroll
        for (int i = 0; i < 10; ++i) {
            const int e = lane + 32 * i;            // covers 0..319 exactly
            dst[e] = (_Float16)(vals[i] * scale);   // e>=300 stores 0 (padding)
        }
        if (lane == 0) {
            if (row < QN) qm[row] = maskv;
            else          dmk[row - QN] = maskv;
        }
    }
    __syncthreads();

    // ---------------- Phase 2: sim = qn @ dn^T via WMMA ----------------
    // 24 tiles (M-tiles: Q=32 -> 2, N-tiles: 192 -> 12); 3 tiles per wave, same M-tile.
    const int sub = lane >> 4;   // half-wave select
    const int nl  = lane & 15;
    const int t0  = w * 3;
    const int mt  = t0 / 12;     // uniform within wave
    const int n0  = t0 % 12;

    v8f acc0 = {}, acc1 = {}, acc2 = {};
    const _Float16* aBase = qA + (mt * 16 + nl) * EPAD;

    #pragma unroll
    for (int kk = 0; kk < EPAD / 32; ++kk) {
        const int kb = kk * 32;
        // A operand: contiguous k order, same as B — lane holds row M=nl,
        // halves 0..15 = K[kb+sub*16 ..] (16 consecutive K)
        union { v16h v; v8h h[2]; } a;
        a.h[0] = *(const v8h*)(aBase + kb + sub * 16);
        a.h[1] = *(const v8h*)(aBase + kb + sub * 16 + 8);
        // B operand: lane holds column N=nl, 16 consecutive K at kb+sub*16
        const v16h b0 = *(const v16h*)(dB + ((n0 + 0) * 16 + nl) * EPAD + kb + sub * 16);
        const v16h b1 = *(const v16h*)(dB + ((n0 + 1) * 16 + nl) * EPAD + kb + sub * 16);
        const v16h b2 = *(const v16h*)(dB + ((n0 + 2) * 16 + nl) * EPAD + kb + sub * 16);
        acc0 = __builtin_amdgcn_wmma_f32_16x16x32_f16(false, a.v, false, b0, (short)0, acc0, false, false);
        acc1 = __builtin_amdgcn_wmma_f32_16x16x32_f16(false, a.v, false, b1, (short)0, acc1, false, false);
        acc2 = __builtin_amdgcn_wmma_f32_16x16x32_f16(false, a.v, false, b2, (short)0, acc2, false, false);
    }

    // ---------------- Phase 3: gaussian kernel pooling (sum over d) ----------------
    // acc element r maps to (m = r + 8*sub, n = nl) within the tile.
    const float dm0 = dmk[(n0 + 0) * 16 + nl];
    const float dm1 = dmk[(n0 + 1) * 16 + nl];
    const float dm2 = dmk[(n0 + 2) * 16 + nl];
    #pragma unroll
    for (int r = 0; r < 8; ++r) {
        const int qrow = mt * 16 + sub * 8 + r;
        const float s0 = acc0[r], s1 = acc1[r], s2 = acc2[r];
        #pragma unroll
        for (int k = 0; k < NK; ++k) {
            const float mu = MU_C[k], c = CF_C[k];
            const float e0 = s0 - mu, e1 = s1 - mu, e2 = s2 - mu;
            const float v = dm0 * __expf(c * e0 * e0)
                          + dm1 * __expf(c * e1 * e1)
                          + dm2 * __expf(c * e2 * e2);
            atomicAdd(&ks[qrow * NK + k], v);
        }
    }
    __syncthreads();

    // ---------------- Phase 4: log, sum over q, dot with w_out ----------------
    if (tid < NK) {
        float acc = 0.0f;
        #pragma unroll
        for (int q = 0; q < QN; ++q) {
            const float v = qm[q] * ks[q * NK + tid];  // q_mask factors out of doc-sum
            acc += __logf(v + 1e-12f);
        }
        part[tid] = acc * w_out[tid];
    }
    __syncthreads();
    if (tid == 0) {
        float o = 0.0f;
        #pragma unroll
        for (int k = 0; k < NK; ++k) o += part[k];
        out[b] = o;
    }
}

extern "C" void kernel_launch(void* const* d_in, const int* in_sizes, int n_in,
                              void* d_out, int out_size, void* d_ws, size_t ws_size,
                              hipStream_t stream) {
    const int*   query = (const int*)d_in[0];    // [B,32]
    const int*   doc   = (const int*)d_in[1];    // [B,180]
    const float* emb   = (const float*)d_in[2];  // [100000,300]
    const float* wout  = (const float*)d_in[3];  // [1,11]
    float* out = (float*)d_out;                  // [B,1]

    const int B = in_sizes[0] / QN;

    const size_t smem = (size_t)(DPAD + QN) * EPAD * 2          // dB + qA (f16)
                      + (size_t)(QN * NK + QN + DPAD + 16) * 4; // ks + qmask + dmask + part
    knrm_fused<<<B, NTHREADS, smem, stream>>>(query, doc, emb, wout, out);
}
